// Transformer5_85237920956546
// MI455X (gfx1250) — compile-verified, matrix operand order checked by reading, not run on hardware
//
#include <hip/hip_runtime.h>
#include <hip/hip_bf16.h>
#include <math.h>

// ---------------- problem constants ----------------
#define N_NODES  50000
#define N_EDGES  800000
#define NODE_DIM 10
#define EDGE_DIM 5
#define HID      64
#define NPAD     50016          // 3126 * 16  (row-tile padded node count)
#define MTILES   3126
#define QKVS_LD  256            // q|k|v|skip concatenated columns

typedef __attribute__((ext_vector_type(16))) _Float16 v16h;
typedef __attribute__((ext_vector_type(8)))  _Float16 v8h;
typedef __attribute__((ext_vector_type(8)))  float    v8f;

// ---------------- helpers ----------------
__device__ inline float atomicMaxF32(float* addr, float val) {
    // IEEE-754 monotone trick: works with -inf initialization
    if (val >= 0.0f)
        return __int_as_float(atomicMax((int*)addr, __float_as_int(val)));
    else
        return __uint_as_float(atomicMin((unsigned int*)addr, __float_as_uint(val)));
}

// ---------------- prep kernels ----------------

// elog[i] = log1p(edge_attr[i]), reused by all 6 layers
__global__ void prep_elog(const float* __restrict__ ea, float* __restrict__ elog, int n) {
    int t = blockIdx.x * blockDim.x + threadIdx.x;
    if (t < n) elog[t] = logf(ea[t] + 1.0f);
}

// layer-0 A matrix: hf16[row*32 + col] = (col<10 && row<N) ? f16(log1p(x)) : 0
__global__ void prep_h0(const float* __restrict__ x, _Float16* __restrict__ hf16) {
    int t = blockIdx.x * blockDim.x + threadIdx.x;
    if (t >= NPAD * 32) return;
    int row = t >> 5, col = t & 31;
    float v = 0.0f;
    if (row < N_NODES && col < NODE_DIM) v = logf(x[row * NODE_DIM + col] + 1.0f);
    hf16[t] = (_Float16)v;
}

// Pre-swizzle all 6 layers' [64 x 256] weight blocks (q|k|v|skip) into the
// per-lane WMMA B-fragment layout:  Bpack[l][kt][nt][lane][i]  (f16).
__global__ void prep_wpack(const float* __restrict__ w1q, const float* __restrict__ w1k,
                           const float* __restrict__ w1v, const float* __restrict__ w1s,
                           const float* __restrict__ wq,  const float* __restrict__ wk,
                           const float* __restrict__ wv,  const float* __restrict__ wsk,
                           _Float16* __restrict__ Bpack) {
    int t = blockIdx.x * blockDim.x + threadIdx.x;
    if (t >= 6 * 64 * 256) return;
    int l = t / (64 * 256);
    int rem = t % (64 * 256);
    int k = rem >> 8;          // 0..63
    int n = rem & 255;         // 0..255
    int c = n & 63;
    float w = 0.0f;
    if (l == 0) {
        if (k < NODE_DIM) {
            const float* W = (n < 64) ? w1q : (n < 128) ? w1k : (n < 192) ? w1v : w1s;
            w = W[k * 64 + c];
        }
    } else {
        int li = l - 1;
        const float* W = (n < 64) ? wq : (n < 128) ? wk : (n < 192) ? wv : wsk;
        w = W[li * 64 * 64 + k * 64 + c];
    }
    // invert the 16-bit A/B fragment K-map (ISA 7.12.2, 16-bit 16x32)
    int kt = k >> 5, kk = k & 31;
    int hi = (kk >> 4) & 1;          // K in [16..31] of the k-tile
    int r  = kk & 15;
    int lane_half = (r >= 8) ? 1 : 0;
    int idx  = (r & 7) + (hi ? 8 : 0);
    int lane = (n & 15) + lane_half * 16;
    int nt   = (n >> 4) & 15;
    Bpack[(size_t)l * 16384 + ((kt * 16 + nt) * 32 + lane) * 16 + idx] = (_Float16)w;
}

// bias concat: bcat[l][0..255] = b_q | b_k | b_v | b_skip
__global__ void prep_bcat(const float* __restrict__ b1q, const float* __restrict__ b1k,
                          const float* __restrict__ b1v, const float* __restrict__ b1s,
                          const float* __restrict__ bq,  const float* __restrict__ bk,
                          const float* __restrict__ bv,  const float* __restrict__ bs,
                          float* __restrict__ bcat) {
    int t = blockIdx.x * blockDim.x + threadIdx.x;
    if (t >= 6 * 256) return;
    int l = t >> 8, cn = t & 255, c = cn & 63;
    float b;
    if (l == 0)
        b = (cn < 64) ? b1q[c] : (cn < 128) ? b1k[c] : (cn < 192) ? b1v[c] : b1s[c];
    else {
        int li = l - 1;
        const float* B = (cn < 64) ? bq : (cn < 128) ? bk : (cn < 192) ? bv : bs;
        b = B[li * 64 + c];
    }
    bcat[t] = b;
}

// ---------------- fused Q|K|V|skip GEMM (WMMA f16 -> f32) ----------------
// grid.x = MTILES (16 node rows each), 256 threads = 8 waves.
// wave w computes output col-tiles nt = w and nt = w+8.
// Templated on LDA / NKT so the WMMA chain is branch-free; all B fragments
// are preloaded into VGPRs before the accumulate phase (one load clause,
// one wait, then back-to-back v_wmma).
template <int LDA, int NKT>
__global__ void __launch_bounds__(256)
gemm_qkvs(const _Float16* __restrict__ A,
          const _Float16* __restrict__ Bp, const float* __restrict__ bias,
          float* __restrict__ out) {
    __shared__ __align__(16) _Float16 sA[16 * 64];

    const int mt   = blockIdx.x;
    const int wave = threadIdx.x >> 5;
    const int lane = threadIdx.x & 31;

    // preload all B fragments for this wave's two col-tiles (global, hits L2/WGP$)
    v16h bfrag[2][NKT];
#pragma unroll
    for (int t = 0; t < 2; ++t) {
        const int nt = wave + t * 8;
#pragma unroll
        for (int kt = 0; kt < NKT; ++kt)
            bfrag[t][kt] = *(const v16h*)(Bp + ((kt * 16 + nt) * 32 + lane) * 16);
    }

    // cooperative stage of the 16 x (32*NKT) A tile into LDS (as u32 words)
    {
        const unsigned int* gsrc = (const unsigned int*)(A + (size_t)mt * 16 * LDA);
        unsigned int* ldst = (unsigned int*)sA;
        const int nwords = 16 * LDA / 2;           // halfs -> dwords
        for (int i = threadIdx.x; i < nwords; i += blockDim.x) ldst[i] = gsrc[i];
    }
    __syncthreads();

    // A fragments (16-bit A 16x32 per-lane layout)
    const int rloc = lane & 15;
    const int b0   = (lane >> 4) * 8;
    v16h afrag[NKT];
#pragma unroll
    for (int kt = 0; kt < NKT; ++kt) {
        const _Float16* ap = sA + rloc * LDA + kt * 32;
        v8h lo = *(const v8h*)(ap + b0);
        v8h hi = *(const v8h*)(ap + 16 + b0);
#pragma unroll
        for (int i = 0; i < 8; ++i) { afrag[kt][i] = lo[i]; afrag[kt][8 + i] = hi[i]; }
    }

    const int rbase = (lane >> 4) * 8;             // C/D: lanes 16..31 hold M+8
#pragma unroll
    for (int t = 0; t < 2; ++t) {
        const int nt  = wave + t * 8;
        const int col = nt * 16 + (lane & 15);
        const float bv = bias[col];
        v8f acc;
#pragma unroll
        for (int r = 0; r < 8; ++r) acc[r] = bv;
#pragma unroll
        for (int kt = 0; kt < NKT; ++kt)
            acc = __builtin_amdgcn_wmma_f32_16x16x32_f16(
                    false, afrag[kt], false, bfrag[t][kt], (short)0, acc, false, false);
        float* op = out + (size_t)(mt * 16 + rbase) * QKVS_LD + col;
#pragma unroll
        for (int r = 0; r < 8; ++r) op[(size_t)r * QKVS_LD] = acc[r];
    }
}

// ---------------- per-layer node state init ----------------
__global__ void init_nodes(float* __restrict__ agg, float* __restrict__ m,
                           float* __restrict__ s) {
    int t = blockIdx.x * blockDim.x + threadIdx.x;
    if (t >= NPAD * HID) return;
    agg[t] = 0.0f;
    if (t < NPAD) { m[t] = -INFINITY; s[t] = 0.0f; }
}

// ---------------- edge pass 1: logits + segment max ----------------
__global__ void __launch_bounds__(256)
edge_logits(const long long* __restrict__ ei, const float* __restrict__ elog,
            const float* __restrict__ we, const float* __restrict__ qkvs,
            float* __restrict__ pbuf, float* __restrict__ mbuf) {
    __shared__ __align__(16) float swe[EDGE_DIM * HID];
    for (int i = threadIdx.x; i < EDGE_DIM * HID; i += blockDim.x) swe[i] = we[i];
    __syncthreads();

    int e = blockIdx.x * blockDim.x + threadIdx.x;
    if (e >= N_EDGES) return;
    int src = (int)ei[e];
    int dst = (int)ei[N_EDGES + e];
    float ea0 = elog[e * 5 + 0], ea1 = elog[e * 5 + 1], ea2 = elog[e * 5 + 2];
    float ea3 = elog[e * 5 + 3], ea4 = elog[e * 5 + 4];
    const float4* q4 = (const float4*)(qkvs + (size_t)dst * QKVS_LD);        // q
    const float4* k4 = (const float4*)(qkvs + (size_t)src * QKVS_LD + 64);   // k
    const float4* w0 = (const float4*)(swe);
    const float4* w1 = (const float4*)(swe + 64);
    const float4* w2 = (const float4*)(swe + 128);
    const float4* w3 = (const float4*)(swe + 192);
    const float4* w4 = (const float4*)(swe + 256);
    float acc = 0.0f;
#pragma unroll 4
    for (int c = 0; c < 16; ++c) {
        float4 q = q4[c], k = k4[c];
        float4 a = w0[c], b = w1[c], d = w2[c], f = w3[c], g = w4[c];
        float e0 = fmaf(ea0, a.x, fmaf(ea1, b.x, fmaf(ea2, d.x, fmaf(ea3, f.x, ea4 * g.x))));
        float e1 = fmaf(ea0, a.y, fmaf(ea1, b.y, fmaf(ea2, d.y, fmaf(ea3, f.y, ea4 * g.y))));
        float e2 = fmaf(ea0, a.z, fmaf(ea1, b.z, fmaf(ea2, d.z, fmaf(ea3, f.z, ea4 * g.z))));
        float e3 = fmaf(ea0, a.w, fmaf(ea1, b.w, fmaf(ea2, d.w, fmaf(ea3, f.w, ea4 * g.w))));
        acc = fmaf(q.x, k.x + e0, acc);
        acc = fmaf(q.y, k.y + e1, acc);
        acc = fmaf(q.z, k.z + e2, acc);
        acc = fmaf(q.w, k.w + e3, acc);
    }
    float l = acc * 0.125f;                                // / sqrt(64)
    pbuf[e] = l;
    atomicMaxF32(&mbuf[dst], l);
}

// ---------------- edge pass 2: exp + segment sum ----------------
__global__ void edge_exp_sum(const long long* __restrict__ ei,
                             float* __restrict__ pbuf, const float* __restrict__ mbuf,
                             float* __restrict__ sbuf) {
    int e = blockIdx.x * blockDim.x + threadIdx.x;
    if (e >= N_EDGES) return;
    int dst = (int)ei[N_EDGES + e];
    float p = __expf(pbuf[e] - mbuf[dst]);
    pbuf[e] = p;
    atomicAdd(&sbuf[dst], p);
}

// ---------------- edge pass 3: weighted value aggregation ----------------
// one wave (32 lanes) per edge, 2 columns per lane
__global__ void __launch_bounds__(256)
edge_aggregate(const long long* __restrict__ ei, const float* __restrict__ elog,
               const float* __restrict__ we, const float* __restrict__ qkvs,
               const float* __restrict__ pbuf, const float* __restrict__ sbuf,
               float* __restrict__ agg) {
    __shared__ __align__(16) float swe[EDGE_DIM * HID];
    for (int i = threadIdx.x; i < EDGE_DIM * HID; i += blockDim.x) swe[i] = we[i];
    __syncthreads();

    int wave = threadIdx.x >> 5, lane = threadIdx.x & 31;
    int e = blockIdx.x * 8 + wave;
    if (e >= N_EDGES) return;
    int src = (int)ei[e];
    int dst = (int)ei[N_EDGES + e];
    float alpha = pbuf[e] / sbuf[dst];
    float ea0 = elog[e * 5 + 0], ea1 = elog[e * 5 + 1], ea2 = elog[e * 5 + 2];
    float ea3 = elog[e * 5 + 3], ea4 = elog[e * 5 + 4];
    const float* vp = qkvs + (size_t)src * QKVS_LD + 128;  // v block
    float* ap = agg + (size_t)dst * HID;
#pragma unroll
    for (int t = 0; t < 2; ++t) {
        int c = lane + t * 32;
        float ee = fmaf(ea0, swe[c],
                   fmaf(ea1, swe[64 + c],
                   fmaf(ea2, swe[128 + c],
                   fmaf(ea3, swe[192 + c], ea4 * swe[256 + c]))));
        atomicAdd(&ap[c], alpha * (vp[c] + ee));
    }
}

// ---------------- node update: h = agg + skip ; emit f32 + f16 ----------------
__global__ void node_update(const float* __restrict__ agg, const float* __restrict__ qkvs,
                            float* __restrict__ h, _Float16* __restrict__ hf16) {
    int t = blockIdx.x * blockDim.x + threadIdx.x;
    if (t >= NPAD * HID) return;
    int row = t >> 6, col = t & 63;
    float v = 0.0f;
    if (row < N_NODES) v = agg[t] + qkvs[(size_t)row * QKVS_LD + 192 + col];
    h[t] = v;
    hf16[t] = (_Float16)v;
}

// ---------------- final linear HID -> 1 ----------------
__global__ void final_linear(const float* __restrict__ h, const float* __restrict__ lw,
                             const float* __restrict__ lb, float* __restrict__ out) {
    int i = blockIdx.x * blockDim.x + threadIdx.x;
    if (i >= N_NODES) return;
    const float4* hp = (const float4*)(h + (size_t)i * HID);
    const float4* wp = (const float4*)lw;
    float acc = lb[0];
#pragma unroll 4
    for (int c = 0; c < 16; ++c) {
        float4 a = hp[c], b = wp[c];
        acc = fmaf(a.x, b.x, acc);
        acc = fmaf(a.y, b.y, acc);
        acc = fmaf(a.z, b.z, acc);
        acc = fmaf(a.w, b.w, acc);
    }
    out[i] = acc;
}

// ---------------- host orchestration ----------------
extern "C" void kernel_launch(void* const* d_in, const int* in_sizes, int n_in,
                              void* d_out, int out_size, void* d_ws, size_t ws_size,
                              hipStream_t stream) {
    const float*     x    = (const float*)d_in[0];
    const long long* ei   = (const long long*)d_in[1];
    const float*     ea   = (const float*)d_in[2];
    const float* w1q = (const float*)d_in[3];  const float* b1q = (const float*)d_in[4];
    const float* w1k = (const float*)d_in[5];  const float* b1k = (const float*)d_in[6];
    const float* w1v = (const float*)d_in[7];  const float* b1v = (const float*)d_in[8];
    const float* w1e = (const float*)d_in[9];
    const float* w1s = (const float*)d_in[10]; const float* b1s = (const float*)d_in[11];
    const float* wq  = (const float*)d_in[12]; const float* bq  = (const float*)d_in[13];
    const float* wk  = (const float*)d_in[14]; const float* bk  = (const float*)d_in[15];
    const float* wv  = (const float*)d_in[16]; const float* bv  = (const float*)d_in[17];
    const float* we  = (const float*)d_in[18];
    const float* wsk = (const float*)d_in[19]; const float* bs  = (const float*)d_in[20];
    const float* lw  = (const float*)d_in[21]; const float* lb  = (const float*)d_in[22];
    float* out = (float*)d_out;

    // workspace carve-up (256B aligned)
    char* ws = (char*)d_ws;
    size_t off = 0;
    auto carve = [&](size_t bytes) { char* p = ws + off; off = (off + bytes + 255) & ~(size_t)255; return p; };
    _Float16* hf16  = (_Float16*)carve((size_t)NPAD * HID * sizeof(_Float16));
    float*    qkvs  = (float*)   carve((size_t)NPAD * QKVS_LD * sizeof(float));
    float*    hbuf  = (float*)   carve((size_t)NPAD * HID * sizeof(float));
    float*    elog  = (float*)   carve((size_t)N_EDGES * EDGE_DIM * sizeof(float));
    float*    pbuf  = (float*)   carve((size_t)N_EDGES * sizeof(float));
    float*    mbuf  = (float*)   carve((size_t)NPAD * sizeof(float));
    float*    sbuf  = (float*)   carve((size_t)NPAD * sizeof(float));
    float*    agg   = (float*)   carve((size_t)NPAD * HID * sizeof(float));
    _Float16* Bpack = (_Float16*)carve((size_t)6 * 64 * 256 * sizeof(_Float16));
    float*    bcat  = (float*)   carve((size_t)6 * 256 * sizeof(float));

    const int B = 256;
    // one-time (per call) preprocessing
    prep_elog<<<(N_EDGES * EDGE_DIM + B - 1) / B, B, 0, stream>>>(ea, elog, N_EDGES * EDGE_DIM);
    prep_h0<<<(NPAD * 32 + B - 1) / B, B, 0, stream>>>(x, hf16);
    prep_wpack<<<(6 * 64 * 256 + B - 1) / B, B, 0, stream>>>(w1q, w1k, w1v, w1s, wq, wk, wv, wsk, Bpack);
    prep_bcat<<<(6 * 256 + B - 1) / B, B, 0, stream>>>(b1q, b1k, b1v, b1s, bq, bk, bv, bs, bcat);

    for (int l = 0; l < 6; ++l) {
        const float* wel = (l == 0) ? w1e : (we + (size_t)(l - 1) * EDGE_DIM * HID);

        if (l == 0)
            gemm_qkvs<32, 1><<<MTILES, B, 0, stream>>>(hf16, Bpack, bcat, qkvs);
        else
            gemm_qkvs<64, 2><<<MTILES, B, 0, stream>>>(hf16, Bpack + (size_t)l * 16384,
                                                       bcat + l * 256, qkvs);
        init_nodes<<<(NPAD * HID + B - 1) / B, B, 0, stream>>>(agg, mbuf, sbuf);
        edge_logits<<<N_EDGES / B, B, 0, stream>>>(ei, elog, wel, qkvs, pbuf, mbuf);
        edge_exp_sum<<<N_EDGES / B, B, 0, stream>>>(ei, pbuf, mbuf, sbuf);
        edge_aggregate<<<N_EDGES / 8, B, 0, stream>>>(ei, elog, wel, qkvs, pbuf, sbuf, agg);
        node_update<<<(NPAD * HID + B - 1) / B, B, 0, stream>>>(agg, qkvs, hbuf, hf16);
    }
    final_linear<<<(N_NODES + B - 1) / B, B, 0, stream>>>(hbuf, lw, lb, out);
}
